// GNNModel_62500364091585
// MI455X (gfx1250) — compile-verified, matrix operand order checked by reading, not run on hardware
//
#include <hip/hip_runtime.h>
#include <hip/hip_bf16.h>

// ---------------------------------------------------------------------------
// GAT (2 layers) for MI455X / gfx1250. wave32, WMMA f32_16x16x32_f16 GEMMs
// with pre-packed f16 weight fragments (B operand = 2x global_load_b128),
// L2-resident scatter-atomic segment softmax + aggregation.
// ---------------------------------------------------------------------------

typedef __attribute__((ext_vector_type(16))) _Float16 v16h;
typedef __attribute__((ext_vector_type(8)))  float    v8f;

static constexpr int NN  = 50000;
static constexpr int EE  = 800000;
static constexpr int ETOT = EE + NN;      // edges + self loops
static constexpr int INC = 128;
static constexpr int HID = 64;
static constexpr int H1H = 4;             // heads layer 1
static constexpr int F1  = H1H * HID;     // 256
static constexpr float NEG_SLOPE = 0.2f;

// ---- workspace float-offset layout (bump allocated, phase-2 overlays H1) ----
// phase 1
static constexpr long long O_H1   = 0;                       // N*256
static constexpr long long O_AS1  = O_H1  + (long long)NN * F1;   // N*4
static constexpr long long O_AD1  = O_AS1 + (long long)NN * H1H;
static constexpr long long O_M1   = O_AD1 + (long long)NN * H1H;  // u32 encoded max
static constexpr long long O_DEN1 = O_M1  + (long long)NN * H1H;
static constexpr long long O_EX1  = O_DEN1+ (long long)NN * H1H;  // ETOT*4
static constexpr long long O_AGG1 = O_EX1 + (long long)ETOT * H1H; // N*256 (becomes relu'd h)
// packed f16 weight fragments (persist whole run)
static constexpr long long O_W1P  = O_AGG1 + (long long)NN * F1;  // 32768 halves
static constexpr long long O_W2P  = O_W1P + 32768 / 2;            // 16384 halves
static constexpr long long WS_FLOATS = O_W2P + 16384 / 2;         // ~119.3MB
// phase 2 (overlays the dead H1 region [0, N*256))
static constexpr long long O_H2   = 0;                        // N*64
static constexpr long long O_AS2  = O_H2  + (long long)NN * HID;
static constexpr long long O_AD2  = O_AS2 + (long long)NN;
static constexpr long long O_M2   = O_AD2 + (long long)NN;
static constexpr long long O_DEN2 = O_M2  + (long long)NN;
static constexpr long long O_EX2  = O_DEN2+ (long long)NN;    // ETOT
static constexpr long long O_AGG2 = O_EX2 + (long long)ETOT;  // N*64

// ---- helpers ---------------------------------------------------------------
__device__ __forceinline__ unsigned fenc(float f) {
    unsigned u = __float_as_uint(f);
    return (u & 0x80000000u) ? ~u : (u | 0x80000000u);
}
__device__ __forceinline__ float fdec(unsigned u) {
    u = (u & 0x80000000u) ? (u & 0x7FFFFFFFu) : ~u;
    return __uint_as_float(u);
}
__device__ __forceinline__ void edge_sd(const int* __restrict__ ei, int e, int& s, int& d) {
    if (e < EE) { s = ei[e]; d = ei[EE + e]; }
    else        { s = e - EE; d = s; }          // self loop
}
__device__ __forceinline__ float lrelu(float v) { return v > 0.f ? v : NEG_SLOPE * v; }

__global__ void zero_f32(float* __restrict__ p, long long n) {
    long long i = (long long)blockIdx.x * blockDim.x + threadIdx.x;
    long long stride = (long long)gridDim.x * blockDim.x;
    for (; i < n; i += stride) p[i] = 0.f;
}

// ---- pack a [nksteps*32, nctiles*16] f32 weight matrix into per-wave WMMA
// B-fragments: frag (ks, ct): lane l holds 16 halves, element e <-> row
// K = ks*32 + (l>=16)*8 + (e<8 ? e : 16+e-8), col = ct*16 + (l&15).
__global__ void pack_w_frag(const float* __restrict__ W, _Float16* __restrict__ Wp,
                            int ncols, int nctiles, int nfrag32) {
    const int idx = blockIdx.x * blockDim.x + threadIdx.x;   // one per (ks,ct,lane)
    if (idx >= nfrag32) return;
    const int lane = idx & 31;
    const int tmp  = idx >> 5;
    const int ct   = tmp % nctiles;
    const int ks   = tmp / nctiles;
    const int l16  = lane & 15;
    const int hiHalf = lane >> 4;
    const int col  = ct * 16 + l16;
    v16h frag;
    #pragma unroll
    for (int e = 0; e < 16; ++e) {
        const int kk = ks * 32 + hiHalf * 8 + (e < 8 ? e : (16 + (e - 8)));
        frag[e] = (_Float16)W[(long long)kk * ncols + col];
    }
    *(v16h*)(Wp + (long long)idx * 16) = frag;
}

// ---- GEMM1: H1[N,256] = f16(X[N,128]) @ W1p, f32 accumulate ----------------
// 4 waves/block; each wave: 16 rows x 64 cols (4 WMMA tiles), K in 32-steps.
__global__ __launch_bounds__(128) void gemm1_wmma(const float* __restrict__ X,
                                                  const _Float16* __restrict__ Wp,
                                                  float* __restrict__ H) {
    const int wave    = threadIdx.x >> 5;
    const int lane    = threadIdx.x & 31;
    const int l16     = lane & 15;
    const int hiHalf  = lane >> 4;              // 0|1 -> K split per ISA layout
    const int rowBase = blockIdx.x * 16;        // 3125 blocks * 16 = 50000 exact
    const int colBase = wave * 64;
    const int row     = rowBase + l16;

    v8f acc[4]; acc[0] = v8f{}; acc[1] = v8f{}; acc[2] = v8f{}; acc[3] = v8f{};

    for (int k = 0; k < INC; k += 32) {
        const int ks    = k >> 5;
        const int kbase = k + hiHalf * 8;
        // A fragment: lane holds K = kbase+0..7 (e=0..7) and kbase+16..23 (e=8..15)
        const float4 x0 = *(const float4*)(X + (long long)row * INC + kbase);
        const float4 x1 = *(const float4*)(X + (long long)row * INC + kbase + 4);
        const float4 x2 = *(const float4*)(X + (long long)row * INC + kbase + 16);
        const float4 x3 = *(const float4*)(X + (long long)row * INC + kbase + 20);
        v16h a;
        a[0]=(_Float16)x0.x; a[1]=(_Float16)x0.y; a[2]=(_Float16)x0.z; a[3]=(_Float16)x0.w;
        a[4]=(_Float16)x1.x; a[5]=(_Float16)x1.y; a[6]=(_Float16)x1.z; a[7]=(_Float16)x1.w;
        a[8]=(_Float16)x2.x; a[9]=(_Float16)x2.y; a[10]=(_Float16)x2.z; a[11]=(_Float16)x2.w;
        a[12]=(_Float16)x3.x; a[13]=(_Float16)x3.y; a[14]=(_Float16)x3.z; a[15]=(_Float16)x3.w;

        #pragma unroll
        for (int t = 0; t < 4; ++t) {
            // packed fragment index: ks*16 colTiles + (wave*4 + t)
            const v16h b = *(const v16h*)(Wp +
                (((long long)ks * 16 + wave * 4 + t) * 32 + lane) * 16);
            acc[t] = __builtin_amdgcn_wmma_f32_16x16x32_f16(
                false, a, false, b, (short)0, acc[t], false, false);
        }
    }

    #pragma unroll
    for (int t = 0; t < 4; ++t) {
        const int col = colBase + t * 16 + l16;
        #pragma unroll
        for (int r = 0; r < 8; ++r) {
            const int rr = rowBase + r + hiHalf * 8;
            H[(long long)rr * F1 + col] = acc[t][r];
        }
    }
}

// ---- GEMM2: H2[N,64] = f16(Hrelu[N,256]) @ W2p -----------------------------
__global__ __launch_bounds__(128) void gemm2_wmma(const float* __restrict__ X,
                                                  const _Float16* __restrict__ Wp,
                                                  float* __restrict__ H) {
    const int wave    = threadIdx.x >> 5;
    const int lane    = threadIdx.x & 31;
    const int l16     = lane & 15;
    const int hiHalf  = lane >> 4;
    const int rowBase = blockIdx.x * 16;
    const int colBase = wave * 16;              // 4 waves cover 64 cols
    const int row     = rowBase + l16;

    v8f acc = v8f{};
    for (int k = 0; k < F1; k += 32) {
        const int ks    = k >> 5;
        const int kbase = k + hiHalf * 8;
        const float4 x0 = *(const float4*)(X + (long long)row * F1 + kbase);
        const float4 x1 = *(const float4*)(X + (long long)row * F1 + kbase + 4);
        const float4 x2 = *(const float4*)(X + (long long)row * F1 + kbase + 16);
        const float4 x3 = *(const float4*)(X + (long long)row * F1 + kbase + 20);
        v16h a;
        a[0]=(_Float16)x0.x; a[1]=(_Float16)x0.y; a[2]=(_Float16)x0.z; a[3]=(_Float16)x0.w;
        a[4]=(_Float16)x1.x; a[5]=(_Float16)x1.y; a[6]=(_Float16)x1.z; a[7]=(_Float16)x1.w;
        a[8]=(_Float16)x2.x; a[9]=(_Float16)x2.y; a[10]=(_Float16)x2.z; a[11]=(_Float16)x2.w;
        a[12]=(_Float16)x3.x; a[13]=(_Float16)x3.y; a[14]=(_Float16)x3.z; a[15]=(_Float16)x3.w;

        const v16h b = *(const v16h*)(Wp +
            (((long long)ks * 4 + wave) * 32 + lane) * 16);
        acc = __builtin_amdgcn_wmma_f32_16x16x32_f16(
            false, a, false, b, (short)0, acc, false, false);
    }

    const int col = colBase + l16;
    #pragma unroll
    for (int r = 0; r < 8; ++r) {
        const int rr = rowBase + r + hiHalf * 8;
        H[(long long)rr * HID + col] = acc[r];
    }
}

// ---- per-node attention coefficients (layer 1: 4 heads) --------------------
__global__ __launch_bounds__(256) void att1_kernel(const float* __restrict__ H1f,
                                                   const float* __restrict__ a_src,
                                                   const float* __restrict__ a_dst,
                                                   float* __restrict__ as,
                                                   float* __restrict__ ad) {
    const int n = blockIdx.x * 256 + threadIdx.x;
    if (n >= NN) return;
    #pragma unroll
    for (int h = 0; h < H1H; ++h) {
        float s = 0.f, d = 0.f;
        #pragma unroll 8
        for (int c = 0; c < HID; ++c) {
            const float v = H1f[(long long)n * F1 + h * HID + c];
            s += v * a_src[h * HID + c];
            d += v * a_dst[h * HID + c];
        }
        as[n * H1H + h] = s;
        ad[n * H1H + h] = d;
    }
}

__global__ __launch_bounds__(256) void att2_kernel(const float* __restrict__ H2f,
                                                   const float* __restrict__ a_src,
                                                   const float* __restrict__ a_dst,
                                                   float* __restrict__ as,
                                                   float* __restrict__ ad) {
    const int n = blockIdx.x * 256 + threadIdx.x;
    if (n >= NN) return;
    float s = 0.f, d = 0.f;
    #pragma unroll 8
    for (int c = 0; c < HID; ++c) {
        const float v = H2f[(long long)n * HID + c];
        s += v * a_src[c];
        d += v * a_dst[c];
    }
    as[n] = s;
    ad[n] = d;
}

// ---- edge kernels: segment max / exp+denom / weighted scatter --------------
__global__ __launch_bounds__(256) void edge_max1(const int* __restrict__ ei,
                                                 const float* __restrict__ as,
                                                 const float* __restrict__ ad,
                                                 unsigned* __restrict__ m) {
    const int e = blockIdx.x * 256 + threadIdx.x;
    if (e >= ETOT) return;
    int s, d; edge_sd(ei, e, s, d);
    #pragma unroll
    for (int h = 0; h < H1H; ++h) {
        const float ev = lrelu(as[s * H1H + h] + ad[d * H1H + h]);
        atomicMax(&m[d * H1H + h], fenc(ev));
    }
}

__global__ __launch_bounds__(256) void edge_den1(const int* __restrict__ ei,
                                                 const float* __restrict__ as,
                                                 const float* __restrict__ ad,
                                                 const unsigned* __restrict__ m,
                                                 float* __restrict__ ex,
                                                 float* __restrict__ den) {
    const int e = blockIdx.x * 256 + threadIdx.x;
    if (e >= ETOT) return;
    int s, d; edge_sd(ei, e, s, d);
    #pragma unroll
    for (int h = 0; h < H1H; ++h) {
        const float ev = lrelu(as[s * H1H + h] + ad[d * H1H + h]);
        const float x = __expf(ev - fdec(m[d * H1H + h]));
        ex[(long long)e * H1H + h] = x;
        atomicAdd(&den[d * H1H + h], x);
    }
}

// one wave32 per edge; lanes split the 4*64 channels
__global__ __launch_bounds__(256) void edge_agg1(const int* __restrict__ ei,
                                                 const float* __restrict__ H1f,
                                                 const float* __restrict__ ex,
                                                 const float* __restrict__ den,
                                                 float* __restrict__ agg) {
    const int e = blockIdx.x * 8 + (threadIdx.x >> 5);
    if (e >= ETOT) return;
    const int lane = threadIdx.x & 31;
    int s, d; edge_sd(ei, e, s, d);
    #pragma unroll
    for (int h = 0; h < H1H; ++h) {
        const float al = ex[(long long)e * H1H + h] / (den[d * H1H + h] + 1e-16f);
        #pragma unroll
        for (int j = 0; j < 2; ++j) {
            const int v = h * HID + j * 32 + lane;
            atomicAdd(&agg[(long long)d * F1 + v], al * H1f[(long long)s * F1 + v]);
        }
    }
}

__global__ __launch_bounds__(256) void edge_max2(const int* __restrict__ ei,
                                                 const float* __restrict__ as,
                                                 const float* __restrict__ ad,
                                                 unsigned* __restrict__ m) {
    const int e = blockIdx.x * 256 + threadIdx.x;
    if (e >= ETOT) return;
    int s, d; edge_sd(ei, e, s, d);
    atomicMax(&m[d], fenc(lrelu(as[s] + ad[d])));
}

__global__ __launch_bounds__(256) void edge_den2(const int* __restrict__ ei,
                                                 const float* __restrict__ as,
                                                 const float* __restrict__ ad,
                                                 const unsigned* __restrict__ m,
                                                 float* __restrict__ ex,
                                                 float* __restrict__ den) {
    const int e = blockIdx.x * 256 + threadIdx.x;
    if (e >= ETOT) return;
    int s, d; edge_sd(ei, e, s, d);
    const float x = __expf(lrelu(as[s] + ad[d]) - fdec(m[d]));
    ex[e] = x;
    atomicAdd(&den[d], x);
}

__global__ __launch_bounds__(256) void edge_agg2(const int* __restrict__ ei,
                                                 const float* __restrict__ H2f,
                                                 const float* __restrict__ ex,
                                                 const float* __restrict__ den,
                                                 float* __restrict__ agg) {
    const int e = blockIdx.x * 8 + (threadIdx.x >> 5);
    if (e >= ETOT) return;
    const int lane = threadIdx.x & 31;
    int s, d; edge_sd(ei, e, s, d);
    const float al = ex[e] / (den[d] + 1e-16f);
    #pragma unroll
    for (int j = 0; j < 2; ++j) {
        const int v = j * 32 + lane;
        atomicAdd(&agg[(long long)d * HID + v], al * H2f[(long long)s * HID + v]);
    }
}

// ---- elementwise: bias + relu in place (layer-1 output -> GEMM2 input) -----
__global__ void bias_relu1(float* __restrict__ agg, const float* __restrict__ b) {
    long long i = (long long)blockIdx.x * blockDim.x + threadIdx.x;
    const long long total = (long long)NN * F1;
    const long long stride = (long long)gridDim.x * blockDim.x;
    for (; i < total; i += stride) {
        const float v = agg[i] + b[i & (F1 - 1)];
        agg[i] = v > 0.f ? v : 0.f;
    }
}

// ---- final: relu(agg2 + b2) @ W_out + b_out -> out[N] ----------------------
__global__ __launch_bounds__(256) void final_out(const float* __restrict__ agg,
                                                 const float* __restrict__ b2,
                                                 const float* __restrict__ Wout,
                                                 const float* __restrict__ bout,
                                                 float* __restrict__ out) {
    const int n = blockIdx.x * 256 + threadIdx.x;
    if (n >= NN) return;
    float acc = bout[0];
    #pragma unroll 8
    for (int c = 0; c < HID; ++c) {
        float v = agg[(long long)n * HID + c] + b2[c];
        v = v > 0.f ? v : 0.f;
        acc += v * Wout[c];
    }
    out[n] = acc;
}

// ---------------------------------------------------------------------------
extern "C" void kernel_launch(void* const* d_in, const int* in_sizes, int n_in,
                              void* d_out, int out_size, void* d_ws, size_t ws_size,
                              hipStream_t stream) {
    (void)in_sizes; (void)n_in; (void)out_size; (void)ws_size;
    const float* x      = (const float*)d_in[0];
    const int*   ei     = (const int*)  d_in[1];
    const float* W1     = (const float*)d_in[2];
    const float* asrc1  = (const float*)d_in[3];
    const float* adst1  = (const float*)d_in[4];
    const float* b1     = (const float*)d_in[5];
    const float* W2     = (const float*)d_in[6];
    const float* asrc2  = (const float*)d_in[7];
    const float* adst2  = (const float*)d_in[8];
    const float* b2     = (const float*)d_in[9];
    const float* Wout   = (const float*)d_in[10];
    const float* bout   = (const float*)d_in[11];
    float* out = (float*)d_out;

    float* ws = (float*)d_ws;
    // phase 1 pointers
    float*    H1   = ws + O_H1;
    float*    as1  = ws + O_AS1;
    float*    ad1  = ws + O_AD1;
    unsigned* m1   = (unsigned*)(ws + O_M1);
    float*    den1 = ws + O_DEN1;
    float*    ex1  = ws + O_EX1;
    float*    agg1 = ws + O_AGG1;
    _Float16* W1p  = (_Float16*)(ws + O_W1P);
    _Float16* W2p  = (_Float16*)(ws + O_W2P);
    // phase 2 pointers (overlay dead H1 region)
    float*    H2   = ws + O_H2;
    float*    as2  = ws + O_AS2;
    float*    ad2  = ws + O_AD2;
    unsigned* m2   = (unsigned*)(ws + O_M2);
    float*    den2 = ws + O_DEN2;
    float*    ex2  = ws + O_EX2;
    float*    agg2 = ws + O_AGG2;

    const int nodeBlocks = (NN + 255) / 256;       // 196
    const int edgeBlocks = (ETOT + 255) / 256;     // 3321
    const int aggBlocks  = (ETOT + 7) / 8;         // 106250
    const int gemmBlocks = NN / 16;                // 3125 exact

    // ---- weight packing (tiny, L2-resident fragments) ----
    pack_w_frag<<<32, 64, 0, stream>>>(W1, W1p, F1, 16, 4 * 16 * 32);   // 128x256
    pack_w_frag<<<16, 64, 0, stream>>>(W2, W2p, HID, 4, 8 * 4 * 32);    // 256x64

    // ---- layer 1 ----
    zero_f32<<<1563, 256, 0, stream>>>((float*)m1, (long long)NN * H1H * 2); // m1+den1 (adjacent)
    zero_f32<<<50000, 256, 0, stream>>>(agg1, (long long)NN * F1);
    gemm1_wmma<<<gemmBlocks, 128, 0, stream>>>(x, W1p, H1);
    att1_kernel<<<nodeBlocks, 256, 0, stream>>>(H1, asrc1, adst1, as1, ad1);
    edge_max1<<<edgeBlocks, 256, 0, stream>>>(ei, as1, ad1, m1);
    edge_den1<<<edgeBlocks, 256, 0, stream>>>(ei, as1, ad1, m1, ex1, den1);
    edge_agg1<<<aggBlocks, 256, 0, stream>>>(ei, H1, ex1, den1, agg1);
    bias_relu1<<<12500, 256, 0, stream>>>(agg1, b1);

    // ---- layer 2 (H1 region now dead; overlay) ----
    zero_f32<<<391, 256, 0, stream>>>((float*)m2, (long long)NN * 2);        // m2+den2 (adjacent)
    zero_f32<<<12500, 256, 0, stream>>>(agg2, (long long)NN * HID);
    gemm2_wmma<<<gemmBlocks, 128, 0, stream>>>(agg1, W2p, H2);
    att2_kernel<<<nodeBlocks, 256, 0, stream>>>(H2, asrc2, adst2, as2, ad2);
    edge_max2<<<edgeBlocks, 256, 0, stream>>>(ei, as2, ad2, m2);
    edge_den2<<<edgeBlocks, 256, 0, stream>>>(ei, as2, ad2, m2, ex2, den2);
    edge_agg2<<<aggBlocks, 256, 0, stream>>>(ei, H2, ex2, den2, agg2);
    final_out<<<nodeBlocks, 256, 0, stream>>>(agg2, b2, Wout, bout, out);
}
